// MyBasicAttention_69020124447188
// MI455X (gfx1250) — matrix operand order repaired and checked by reading, not compiled, not run on hardware
//
#include <hip/hip_runtime.h>
#include <hip/hip_bf16.h>
#include <math.h>

// ---------------------------------------------------------------------------
// MI455X (gfx1250) fused attention:
//   scores = xs @ ys^T ; mask ; softmax ; emb = weight @ ys
//   weight (softmax probs) is a required output -> materialize it, but keep
//   the 32x1024 score strip in LDS between GEMM1 and GEMM2 (no HBM round trip).
//   bf16 hi/lo split (3 WMMAs) recovers ~fp32 precision on the bf16 matrix pipe.
// ---------------------------------------------------------------------------

typedef __attribute__((ext_vector_type(16))) __bf16 v16bf;
typedef __attribute__((ext_vector_type(8)))  __bf16 v8bf;
typedef __attribute__((ext_vector_type(8)))  float  v8f;

constexpr int BATCH = 16;
constexpr int XL    = 1024;
constexpr int YL    = 1024;
constexpr int DIM   = 1024;
constexpr int BM    = 32;          // query rows per workgroup
constexpr int PITCH = 1032;        // LDS row pitch (floats), pad to dodge bank conflicts
constexpr float NEG_INF = -1e20f;

#define WMMA_BF16(A, Bm, C) \
  __builtin_amdgcn_wmma_f32_16x16x32_bf16(false, (A), false, (Bm), (short)0, (C), false, false)

// ---- fragment loaders (match CDNA5 16-bit WMMA VGPR layouts) ---------------
// A (16x32): lane<16 -> M=lane, K runs [0..7] & [16..23]; lane>=16 -> K runs [8..15] & [24..31]
__device__ __forceinline__ v16bf load_fragA(const __bf16* p, int ld, int lr, int hf) {
  union { v16bf v; v8bf h[2]; } f;
  const __bf16* q = p + (size_t)lr * ld + hf * 8;
  f.h[0] = *(const v8bf*)(q);
  f.h[1] = *(const v8bf*)(q + 16);
  return f.v;
}
// B (32x16): lane<16 -> N=lane, K runs [0..7] & [16..23]; lane>=16 -> K runs [8..15] & [24..31]
__device__ __forceinline__ v16bf load_fragB(const __bf16* p, int ld, int lr, int hf) {
  union { v16bf v; v8bf h[2]; } f;
  const __bf16* q = p + (size_t)lr * ld + hf * 8;
  f.h[0] = *(const v8bf*)(q);
  f.h[1] = *(const v8bf*)(q + 16);
  return f.v;
}
// A fragment from fp32 LDS with on-the-fly hi/lo bf16 split
__device__ __forceinline__ void load_fragA_f32(const float* p, int ld, int lr, int hf,
                                               v16bf& hi, v16bf& lo) {
  union U { v16bf v; __bf16 e[16]; } H, L;
  const float* q = p + (size_t)lr * ld + hf * 8;
#pragma unroll
  for (int j = 0; j < 8; ++j) {
    float x0 = q[j];
    __bf16 h0 = (__bf16)x0; H.e[j]     = h0; L.e[j]     = (__bf16)(x0 - (float)h0);
    float x1 = q[16 + j];
    __bf16 h1 = (__bf16)x1; H.e[8 + j] = h1; L.e[8 + j] = (__bf16)(x1 - (float)h1);
  }
  hi = H.v; lo = L.v;
}

// ---- pre-pass: fp32 -> bf16 hi/lo split ------------------------------------
__global__ void cvt_split_kernel(const float* __restrict__ x,
                                 __bf16* __restrict__ hi, __bf16* __restrict__ lo,
                                 unsigned n) {
  unsigned i = blockIdx.x * blockDim.x + threadIdx.x;
  if (i < n) {
    float v = x[i];
    __bf16 h = (__bf16)v;
    hi[i] = h;
    lo[i] = (__bf16)(v - (float)h);
  }
}

// ---- pre-pass: ysT[b,d,y] = ys[b,y,d] as bf16 hi/lo ------------------------
__global__ void transpose_split_kernel(const float* __restrict__ ys,
                                       __bf16* __restrict__ tH, __bf16* __restrict__ tL) {
  __shared__ float tile[32][33];
  const int dT  = DIM / 32;
  const int bid = blockIdx.x;
  const int b   = bid / (32 * dT);
  const int rem = bid % (32 * dT);
  const int y0  = (rem / dT) * 32;
  const int d0  = (rem % dT) * 32;
  const int tx  = threadIdx.x & 31;
  const int ty0 = threadIdx.x >> 5;
#pragma unroll
  for (int ty = ty0; ty < 32; ty += 8)
    tile[ty][tx] = ys[((size_t)(b * YL + y0 + ty)) * DIM + d0 + tx];
  __syncthreads();
#pragma unroll
  for (int ty = ty0; ty < 32; ty += 8) {
    float x = tile[tx][ty];
    __bf16 h = (__bf16)x;
    size_t o = ((size_t)(b * DIM + d0 + ty)) * YL + y0 + tx;
    tH[o] = h;
    tL[o] = (__bf16)(x - (float)h);
  }
}

// ---- fused attention: GEMM1 + mask + softmax + GEMM2 -----------------------
__global__ __launch_bounds__(256)
void attn_fused_kernel(const __bf16* __restrict__ xs_hi, const __bf16* __restrict__ xs_lo,
                       const __bf16* __restrict__ ys_hi, const __bf16* __restrict__ ys_lo,
                       const __bf16* __restrict__ ysT_hi, const __bf16* __restrict__ ysT_lo,
                       const int* __restrict__ mask,
                       float* __restrict__ emb, float* __restrict__ wout) {
  extern __shared__ char smem_raw[];
  float* sc     = (float*)smem_raw;        // [BM][PITCH] scores -> exp -> probs
  float* red    = sc + BM * PITCH;         // [BM][8] partial reductions
  float* rowmax = red + BM * 8;            // [BM]
  float* rowsum = rowmax + BM;             // [BM]

  const int tid  = threadIdx.x;
  const int lane = tid & 31;
  const int wid  = tid >> 5;
  const int wm   = wid & 1;                // wave row-strip  (2 x 16 rows)
  const int wn   = wid >> 1;               // wave col-group  (4 x 256 cols)
  const int hf   = (lane >> 4) & 1;
  const int lr   = lane & 15;

  const int blocksPerB = XL / BM;
  const int b  = blockIdx.x / blocksPerB;
  const int m0 = (blockIdx.x % blocksPerB) * BM;

  v8f acc[16];
#pragma unroll
  for (int t = 0; t < 16; ++t) acc[t] = v8f{};

  // ---------------- GEMM1: scores = xs @ ys^T (bf16 hi/lo, 3-term) ----------
  const __bf16* xh = xs_hi + ((size_t)(b * XL + m0 + wm * 16)) * DIM;
  const __bf16* xl = xs_lo + ((size_t)(b * XL + m0 + wm * 16)) * DIM;
  const int nbase = wn * 256;

  for (int k0 = 0; k0 < DIM; k0 += 32) {
    __builtin_prefetch(xh + k0 + 32, 0, 1);
    v16bf a_h = load_fragA(xh + k0, DIM, lr, hf);
    v16bf a_l = load_fragA(xl + k0, DIM, lr, hf);
#pragma unroll
    for (int t = 0; t < 16; ++t) {
      const size_t yb = ((size_t)(b * YL + nbase + t * 16)) * DIM + k0;
      v16bf b_h = load_fragB(ys_hi + yb, DIM, lr, hf);
      v16bf b_l = load_fragB(ys_lo + yb, DIM, lr, hf);
      acc[t] = WMMA_BF16(a_h, b_h, acc[t]);
      acc[t] = WMMA_BF16(a_h, b_l, acc[t]);
      acc[t] = WMMA_BF16(a_l, b_h, acc[t]);
    }
  }

  // ---------------- mask + spill scores to LDS ------------------------------
#pragma unroll
  for (int t = 0; t < 16; ++t) {
    const int col = nbase + t * 16 + lr;
    const int mk  = mask[b * YL + col];
#pragma unroll
    for (int r = 0; r < 8; ++r) {
      const int row = wm * 16 + hf * 8 + r;
      sc[row * PITCH + col] = mk ? acc[t][r] : NEG_INF;
    }
  }
  __syncthreads();

  // ---------------- softmax over y (row-wise, 8 threads / row) --------------
  const int srow = tid >> 3;
  const int ssub = tid & 7;
  {
    float m = -3.0e38f;
    for (int c = ssub * 128; c < ssub * 128 + 128; ++c)
      m = fmaxf(m, sc[srow * PITCH + c]);
    red[srow * 8 + ssub] = m;
  }
  __syncthreads();
  if (ssub == 0) {
    float m = red[srow * 8];
#pragma unroll
    for (int j = 1; j < 8; ++j) m = fmaxf(m, red[srow * 8 + j]);
    rowmax[srow] = m;
  }
  __syncthreads();
  {
    const float rm = rowmax[srow];
    float s = 0.f;
    for (int c = ssub * 128; c < ssub * 128 + 128; ++c) {
      float e = __expf(sc[srow * PITCH + c] - rm);   // masked -> exp(-huge) = 0
      sc[srow * PITCH + c] = e;
      s += e;
    }
    red[srow * 8 + ssub] = s;
  }
  __syncthreads();
  if (ssub == 0) {
    float s = 0.f;
#pragma unroll
    for (int j = 0; j < 8; ++j) s += red[srow * 8 + j];
    rowsum[srow] = s;
  }
  __syncthreads();

  // normalize in LDS + write weight output (coalesced)
  for (int idx = tid; idx < BM * YL; idx += 256) {
    const int row = idx >> 10;
    const int col = idx & (YL - 1);
    const float p = sc[row * PITCH + col] * (1.0f / rowsum[row]);
    sc[row * PITCH + col] = p;
    wout[((size_t)(b * XL + m0 + row)) * YL + col] = p;
  }
  __syncthreads();

  // ---------------- GEMM2: emb = P @ ys  (B from transposed ysT) ------------
#pragma unroll
  for (int t = 0; t < 16; ++t) acc[t] = v8f{};
  const int dbase = wn * 256;
  const float* Pbase = sc + (size_t)(wm * 16) * PITCH;

  for (int ky = 0; ky < YL; ky += 32) {
    v16bf a_h, a_l;
    load_fragA_f32(Pbase + ky, PITCH, lr, hf, a_h, a_l);
#pragma unroll
    for (int t = 0; t < 16; ++t) {
      const size_t tb = ((size_t)(b * DIM + dbase + t * 16)) * YL + ky;
      v16bf b_h = load_fragB(ysT_hi + tb, YL, lr, hf);
      v16bf b_l = load_fragB(ysT_lo + tb, YL, lr, hf);
      acc[t] = WMMA_BF16(a_h, b_h, acc[t]);
      acc[t] = WMMA_BF16(a_h, b_l, acc[t]);
      acc[t] = WMMA_BF16(a_l, b_h, acc[t]);
    }
  }

#pragma unroll
  for (int t = 0; t < 16; ++t) {
#pragma unroll
    for (int r = 0; r < 8; ++r) {
      const int row = wm * 16 + hf * 8 + r;
      emb[((size_t)(b * XL + m0 + row)) * DIM + dbase + t * 16 + lr] = acc[t][r];
    }
  }
}

// ---------------------------------------------------------------------------
extern "C" void kernel_launch(void* const* d_in, const int* in_sizes, int n_in,
                              void* d_out, int out_size, void* d_ws, size_t ws_size,
                              hipStream_t stream) {
  const float* xs   = (const float*)d_in[0];
  const float* ys   = (const float*)d_in[1];
  const int*   mask = (const int*)d_in[2];

  float* emb  = (float*)d_out;                                   // [B, XL, D]
  float* wout = (float*)d_out + (size_t)BATCH * XL * DIM;        // [B, XL, YL]

  const size_t N = (size_t)BATCH * XL * DIM;                     // 16M elements
  __bf16* ws     = (__bf16*)d_ws;
  __bf16* xs_hi  = ws;
  __bf16* xs_lo  = ws + N;
  __bf16* ys_hi  = ws + 2 * N;
  __bf16* ys_lo  = ws + 3 * N;
  __bf16* ysT_hi = ws + 4 * N;
  __bf16* ysT_lo = ws + 5 * N;

  const unsigned nU = (unsigned)N;
  cvt_split_kernel<<<(nU + 255) / 256, 256, 0, stream>>>(xs, xs_hi, xs_lo, nU);
  cvt_split_kernel<<<(nU + 255) / 256, 256, 0, stream>>>(ys, ys_hi, ys_lo, nU);
  transpose_split_kernel<<<BATCH * (YL / 32) * (DIM / 32), 256, 0, stream>>>(ys, ysT_hi, ysT_lo);

  const size_t shmem = (size_t)(BM * PITCH + BM * 8 + BM + BM) * sizeof(float); // ~132 KB of 320 KB LDS
  attn_fused_kernel<<<BATCH * (XL / BM), 256, shmem, stream>>>(
      xs_hi, xs_lo, ys_hi, ys_lo, ysT_hi, ysT_lo, mask, emb, wout);
}
